// DecomposableAttention_86131274154667
// MI455X (gfx1250) — matrix operand order repaired and checked by reading, not compiled, not run on hardware
//
#include <hip/hip_runtime.h>

// ---------------------------------------------------------------------------
// Decomposable Attention on MI455X (gfx1250)
//   - all GEMMs via v_wmma_f32_16x16x32_bf16 (wave32, 16x16x32 tiles)
//   - LDS staging via Tensor Data Mover (tensor_load_to_lds) when available,
//     double buffered, synced with s_wait_tensorcnt + barrier
// ---------------------------------------------------------------------------

typedef __bf16 bf16;
typedef __attribute__((ext_vector_type(8)))  __bf16 bf16x8;
typedef __attribute__((ext_vector_type(16))) __bf16 v16bf;
typedef __attribute__((ext_vector_type(8)))  float  v8f;

#define SHUF16(lo, hi) __builtin_shufflevector(lo, hi, 0,1,2,3,4,5,6,7,8,9,10,11,12,13,14,15)

static constexpr int AP = 40;           // LDS row pitch (bf16) = 32 data + 8 pad
static constexpr int BBATCH = 512;      // B
static constexpr int LSEQ   = 256;      // L
static constexpr int EMB    = 300, EMBP = 320;
static constexpr int HID    = 200, HIDP = 224;
static constexpr int AGGK   = 400, AGGKP = 416;
static constexpr long long ML = (long long)BBATCH * LSEQ;   // 131072 rows

#if __has_builtin(__builtin_amdgcn_tensor_load_to_lds) && __has_builtin(__builtin_amdgcn_s_wait_tensorcnt)
#define HAVE_TDM 1
#else
#define HAVE_TDM 0
#endif

#if HAVE_TDM
typedef unsigned int v4u __attribute__((ext_vector_type(4)));
typedef int          v4i __attribute__((ext_vector_type(4)));
typedef int          v8i __attribute__((ext_vector_type(8)));

// Generic flat pointer to LDS: hardware takes addr[31:0] as the LDS byte offset
__device__ __forceinline__ unsigned lds_addr_of(const void* p) {
    return (unsigned)(unsigned long long)p;
}

// Issue one TDM 2D tile load: tileK x tileM bf16 tile, row stride strideElems,
// LDS destination padded 16 DWORDs data + 4 DWORDs pad  ->  pitch 40 bf16 (== AP)
// D# layout per CDNA5 ISA 8.3 / 8.4.
__device__ __forceinline__ void tdm_load_tile_2d(unsigned lds_byte_addr, const void* gsrc,
                                                 unsigned tileK, unsigned tileM,
                                                 unsigned strideElems)
{
    unsigned long long ga = (unsigned long long)gsrc;
    v4u g0;
    g0.x = 1u;                                               // count=1, user mode
    g0.y = lds_byte_addr;                                    // lds_addr
    g0.z = (unsigned)ga;                                     // global_addr[31:0]
    g0.w = (unsigned)((ga >> 32) & 0x1FFFFFFull) | (2u << 30); // global_addr[56:32] | type=2
    v8i g1;
    g1[0] = (int)((1u << 16)     // data_size = 1 -> 2 bytes
                | (1u << 20)     // pad_enable
                | (3u << 22)     // pad_interval code 3 = 16 DWORDs (one 32-elem bf16 row)
                | (3u << 25));   // pad_amount  code 3 =  4 DWORDs (8 bf16)  -> pitch 40
    g1[1] = (int)((tileK & 0xFFFFu) << 16);                  // tensor_dim0[15:0]
    g1[2] = (int)((tileK >> 16) | ((tileM & 0xFFFFu) << 16));// tensor_dim0[31:16] | tensor_dim1[15:0]
    g1[3] = (int)((tileM >> 16) | ((tileK & 0xFFFFu) << 16));// tensor_dim1[31:16] | tile_dim0
    g1[4] = (int)(tileM & 0xFFFFu);                          // tile_dim1 (tile_dim2 = 0)
    g1[5] = (int)strideElems;                                // tensor_dim0_stride[31:0]
    g1[6] = 0;                                               // stride hi | dim1_stride lo
    g1[7] = 0;
    v4i z4 = {0, 0, 0, 0};
#if __clang_major__ >= 23
    v8i z8 = {0, 0, 0, 0, 0, 0, 0, 0};
    __builtin_amdgcn_tensor_load_to_lds(g0, g1, z4, z4, z8, 0);
#else
    __builtin_amdgcn_tensor_load_to_lds(g0, g1, z4, z4, 0);
#endif
}
#endif // HAVE_TDM

// ---------------------------------------------------------------------------
// Generic tiled WMMA GEMM:  C[M x ldc] (+= / =) op( A[M x K] * B )  (+bias,ReLU)
//   A: row-major bf16, lda (K padded to mult of 32, zero-filled)
//   B: if BT: pre-transposed N x K row-major bf16 (ldb = K pitch)  [TDM path]
//      else : K x N row-major bf16 (ldb = N pitch), transposed into LDS on load
//   Block tile 128x64, 8 waves (wave32), wave tile 32x32 = 2x2 wmma accumulators
// ---------------------------------------------------------------------------
template<bool OUTF32, bool RELU, bool BIAS, bool ACC, bool BT>
__global__ __launch_bounds__(256)
void gemm_wmma(const bf16* __restrict__ A,  long long sA,
               const bf16* __restrict__ Bm, long long sB,
               void* __restrict__ Cv,       long long sC,
               const float* __restrict__ Cin,
               const float* __restrict__ bias, int Nbias,
               int K, int lda, int ldb, int ldc)
{
    __shared__ bf16 As[2][128 * AP];
    __shared__ bf16 Bs[2][64 * AP];

    const int z = blockIdx.z;
    A  += (size_t)z * sA;
    Bm += (size_t)z * sB;
    float* Cf = (float*)Cv + (size_t)z * sC;
    bf16*  Ch = (bf16*)Cv  + (size_t)z * sC;

    const int m0 = blockIdx.x * 128;
    const int n0 = blockIdx.y * 64;

    const int tid  = threadIdx.x;
    const int lane = tid & 31;
    const int wave = tid >> 5;
    const int wm   = wave & 3;          // wave row (4 in M)
    const int wn   = wave >> 2;         // wave col (2 in N)

    const int arow = tid >> 2;          // 0..63
    const int acol = (tid & 3) * 8;     // 0,8,16,24
    const int kr   = tid >> 3;          // 0..31   (BT=false)
    const int nc   = (tid & 7) * 8;     // 0..56   (BT=false)

    v8f acc[2][2] = {};
    const int nk = K >> 5;
    int cur = 0;

    // ---- manual loaders (fallback + BT=false B path) ----
    auto ldA = [&](int kt, uint4& a0, uint4& a1) {
        const bf16* pa = A + (size_t)(m0 + arow) * lda + kt * 32 + acol;
        a0 = *(const uint4*)pa;
        a1 = *(const uint4*)(pa + (size_t)64 * lda);
    };
    auto stA = [&](int buf, const uint4& a0, const uint4& a1) {
        *(uint4*)&As[buf][arow * AP + acol]        = a0;
        *(uint4*)&As[buf][(arow + 64) * AP + acol] = a1;
    };
    auto ldBt = [&](int kt, uint4& b0) {
        b0 = *(const uint4*)(Bm + (size_t)(n0 + arow) * ldb + kt * 32 + acol);
    };
    auto stBt = [&](int buf, const uint4& b0) {
        *(uint4*)&Bs[buf][arow * AP + acol] = b0;
    };
    auto ldB = [&](int kt, bf16x8& bv) {
        bv = *(const bf16x8*)(Bm + (size_t)(kt * 32 + kr) * ldb + n0 + nc);
    };
    auto stB = [&](int buf, const bf16x8& bv) {
        #pragma unroll
        for (int i = 0; i < 8; ++i) Bs[buf][(nc + i) * AP + kr] = bv[i];
    };

    // ---- compute one K-step from LDS buffer ----
    auto compute = [&](int buf) {
        v16bf af[2], bfv[2];
        #pragma unroll
        for (int mi = 0; mi < 2; ++mi) {
            // A 16x32: lanes 0-15 row M, K{0..7,16..23}; lanes 16-31: K{8..15,24..31}
            const bf16* p = &As[buf][(wm * 32 + mi * 16 + (lane & 15)) * AP + ((lane >> 4) * 8)];
            bf16x8 lo = *(const bf16x8*)p;
            bf16x8 hi = *(const bf16x8*)(p + 16);
            af[mi] = SHUF16(lo, hi);
        }
        #pragma unroll
        for (int ni = 0; ni < 2; ++ni) {
            // B 32x16: lanes 0-15 col N K{0..7,16..23}; lanes 16-31 col N K{8..15,24..31}
            const bf16* p = &Bs[buf][(wn * 32 + ni * 16 + (lane & 15)) * AP + ((lane >> 4) * 8)];
            bf16x8 lo = *(const bf16x8*)p;
            bf16x8 hi = *(const bf16x8*)(p + 16);
            bfv[ni] = SHUF16(lo, hi);
        }
        #pragma unroll
        for (int mi = 0; mi < 2; ++mi)
            #pragma unroll
            for (int ni = 0; ni < 2; ++ni)
                acc[mi][ni] = __builtin_amdgcn_wmma_f32_16x16x32_bf16(
                    false, af[mi], false, bfv[ni], (short)0, acc[mi][ni], false, false);
    };

#if HAVE_TDM
    // ========== TDM (async tensor) staging path ==========
    {
        bf16x8 rbv;
        const int lastTdmWave = BT ? 1 : 0;
        if (wave == 0)
            tdm_load_tile_2d(lds_addr_of(&As[0][0]), A + (size_t)m0 * lda, 32, 128, (unsigned)lda);
        if constexpr (BT) {
            if (wave == 1)
                tdm_load_tile_2d(lds_addr_of(&Bs[0][0]), Bm + (size_t)n0 * ldb, 32, 64, (unsigned)ldb);
        } else {
            ldB(0, rbv);
            stB(0, rbv);
        }
        if (wave <= lastTdmWave) __builtin_amdgcn_s_wait_tensorcnt(0);
        __syncthreads();

        for (int kt = 0; kt < nk; ++kt) {
            const bool nxt = (kt + 1) < nk;
            if (nxt) {
                if (wave == 0)
                    tdm_load_tile_2d(lds_addr_of(&As[cur ^ 1][0]),
                                     A + (size_t)m0 * lda + (kt + 1) * 32, 32, 128, (unsigned)lda);
                if constexpr (BT) {
                    if (wave == 1)
                        tdm_load_tile_2d(lds_addr_of(&Bs[cur ^ 1][0]),
                                         Bm + (size_t)n0 * ldb + (kt + 1) * 32, 32, 64, (unsigned)ldb);
                } else {
                    ldB(kt + 1, rbv);
                }
            }
            compute(cur);
            if (nxt) {
                if constexpr (!BT) stB(cur ^ 1, rbv);
                if (wave <= lastTdmWave) __builtin_amdgcn_s_wait_tensorcnt(0);
            }
            __syncthreads();
            cur ^= 1;
        }
    }
#else
    // ========== manual double-buffered staging (fallback) ==========
    {
        uint4 ra0, ra1, rb0;
        bf16x8 rbv;
        ldA(0, ra0, ra1);
        stA(0, ra0, ra1);
        if constexpr (BT) { ldBt(0, rb0); stBt(0, rb0); }
        else              { ldB(0, rbv);  stB(0, rbv); }
        __syncthreads();

        for (int kt = 0; kt < nk; ++kt) {
            const bool nxt = (kt + 1) < nk;
            if (nxt) {
                ldA(kt + 1, ra0, ra1);
                if constexpr (BT) ldBt(kt + 1, rb0);
                else              ldB(kt + 1, rbv);
            }
            if (kt + 2 < nk)
                __builtin_prefetch(A + (size_t)(m0 + arow) * lda + (kt + 2) * 32 + acol, 0, 1);
            compute(cur);
            if (nxt) {
                stA(cur ^ 1, ra0, ra1);
                if constexpr (BT) stBt(cur ^ 1, rb0);
                else              stB(cur ^ 1, rbv);
            }
            __syncthreads();
            cur ^= 1;
        }
    }
#endif

    // --- epilogue: C layout lanes 0-15 -> M = r, lanes 16-31 -> M = r+8, N = lane&15 ---
    #pragma unroll
    for (int mi = 0; mi < 2; ++mi) {
        #pragma unroll
        for (int ni = 0; ni < 2; ++ni) {
            const int gm0 = m0 + wm * 32 + mi * 16 + ((lane >> 4) * 8);
            const int gn  = n0 + wn * 32 + ni * 16 + (lane & 15);
            if (gn < ldc) {
                float bv = 0.0f;
                if constexpr (BIAS) bv = (gn < Nbias) ? bias[gn] : 0.0f;
                #pragma unroll
                for (int r = 0; r < 8; ++r) {
                    const size_t off = (size_t)(gm0 + r) * ldc + gn;
                    float v = acc[mi][ni][r] + bv;
                    if constexpr (ACC) v += Cin[off];
                    if constexpr (RELU) v = fmaxf(v, 0.0f);
                    if constexpr (OUTF32) Cf[off] = v; else Ch[off] = (bf16)v;
                }
            }
        }
    }
}

// ---------------------------------------------------------------------------
// Weight convert: fp32 (K x N) -> bf16 transposed (Npad x Kpad), zero padded
// ---------------------------------------------------------------------------
__global__ void conv_weight(const float* __restrict__ src, int rowOff, int K, int N,
                            bf16* __restrict__ dst, int Kpad, int Npad)
{
    const long long idx = (long long)blockIdx.x * 256 + threadIdx.x;
    if (idx >= (long long)Kpad * Npad) return;
    const int n = (int)(idx / Kpad);
    const int k = (int)(idx % Kpad);
    float v = (n < N && k < K) ? src[(size_t)(k + rowOff) * N + n] : 0.0f;
    dst[idx] = (bf16)v;
}

// ---------------------------------------------------------------------------
// Embedding gather -> bf16, K padded 300 -> 320 (zero)
// ---------------------------------------------------------------------------
__global__ void gather_embed(const float* __restrict__ emb, const int* __restrict__ s,
                             bf16* __restrict__ dst)
{
    const long long row = blockIdx.x;                 // 0 .. B*L-1
    const int tok = s[row];
    const float* src = emb + (size_t)tok * EMB;
    for (int c = threadIdx.x; c < EMBP; c += blockDim.x)
        dst[row * EMBP + c] = (c < EMB) ? (bf16)src[c] : (bf16)0.0f;
}

// ---------------------------------------------------------------------------
// Softmax helpers (faithful to reference's axis-2 max reused for both norms)
// ---------------------------------------------------------------------------
__global__ void row_max_sum(const float* __restrict__ e, const int* __restrict__ len2,
                            float* __restrict__ rm, float* __restrict__ sb)
{
    __shared__ float red[256];
    const long long row = blockIdx.x;                 // (b, i)
    const int b = (int)(row >> 8);
    const int j = threadIdx.x;
    const float v = e[row * 256 + j];
    red[j] = v; __syncthreads();
    for (int s = 128; s > 0; s >>= 1) { if (j < s) red[j] = fmaxf(red[j], red[j + s]); __syncthreads(); }
    const float m = red[0]; __syncthreads();
    const float ex = (j < len2[b]) ? __expf(v - m) : 0.0f;
    red[j] = ex; __syncthreads();
    for (int s = 128; s > 0; s >>= 1) { if (j < s) red[j] += red[j + s]; __syncthreads(); }
    if (j == 0) { rm[row] = m; sb[row] = red[0]; }
}

__global__ void col_sum(const float* __restrict__ e, const float* __restrict__ rm,
                        const int* __restrict__ len1, float* __restrict__ sa)
{
    __shared__ float red[256];
    const long long col = blockIdx.x;                 // (b, j)
    const int b = (int)(col >> 8);
    const int j = (int)(col & 255);
    const int i = threadIdx.x;
    const float m = rm[col];                          // quirk: rowmax of row j
    const float v = e[((size_t)b * 256 + i) * 256 + j];
    const float ex = (i < len1[b]) ? __expf(v - m) : 0.0f;
    red[i] = ex; __syncthreads();
    for (int s = 128; s > 0; s >>= 1) { if (i < s) red[i] += red[i + s]; __syncthreads(); }
    if (i == 0) sa[col] = red[0];
}

__global__ void norm_write(const float* __restrict__ e, const float* __restrict__ rm,
                           const float* __restrict__ sa, const float* __restrict__ sb,
                           const int* __restrict__ len1, const int* __restrict__ len2,
                           bf16* __restrict__ e_beta, bf16* __restrict__ e_alphaT)
{
    const long long gid = (long long)blockIdx.x * 256 + threadIdx.x;  // b*65536 + i*256 + j
    const int b = (int)(gid >> 16);
    const int i = (int)((gid >> 8) & 255);
    const int j = (int)(gid & 255);
    const float x = e[gid];
    const float beta  = (j < len2[b]) ? __expf(x - rm[(size_t)b * 256 + i]) / sb[(size_t)b * 256 + i] : 0.0f;
    const float alpha = (i < len1[b]) ? __expf(x - rm[(size_t)b * 256 + j]) / sa[(size_t)b * 256 + j] : 0.0f;
    e_beta[gid] = (bf16)beta;
    e_alphaT[((size_t)b << 16) + (size_t)j * 256 + i] = (bf16)alpha;   // store transposed
}

// ---------------------------------------------------------------------------
// Masked sum over L:  dst[b][dstOff+n] = sum_{l<len[b]} X[(b*L+l)*srcLd + n]
// ---------------------------------------------------------------------------
__global__ void masked_sum(const bf16* __restrict__ X, const int* __restrict__ len,
                           bf16* __restrict__ dst, int dstLd, int dstOff, int Ncopy, int srcLd)
{
    const int b = blockIdx.x;
    const int n = threadIdx.x;
    if (n >= Ncopy) return;
    const int Lb = len[b];
    float s = 0.0f;
    for (int l = 0; l < Lb; ++l)
        s += (float)X[((size_t)b * LSEQ + l) * srcLd + n];
    dst[(size_t)b * dstLd + dstOff + n] = (bf16)s;
}

// ---------------------------------------------------------------------------
// Final tiny GEMM: out[b][j] = b2g[j] + sum_k aggH[b][k] * W2g[k][j],  j in {0,1}
// ---------------------------------------------------------------------------
__global__ void final_out(const bf16* __restrict__ H, const float* __restrict__ W2g,
                          const float* __restrict__ b2g, float* __restrict__ out)
{
    const int idx = blockIdx.x * blockDim.x + threadIdx.x;
    if (idx >= BBATCH * 2) return;
    const int b = idx >> 1, j = idx & 1;
    float s = b2g[j];
    for (int k = 0; k < HID; ++k)
        s += (float)H[(size_t)b * HIDP + k] * W2g[(size_t)k * 2 + j];
    out[idx] = s;
}

// ---------------------------------------------------------------------------
extern "C" void kernel_launch(void* const* d_in, const int* in_sizes, int n_in,
                              void* d_out, int out_size, void* d_ws, size_t ws_size,
                              hipStream_t stream)
{
    const float* emb = (const float*)d_in[0];
    const float* W1a = (const float*)d_in[1];  const float* b1a = (const float*)d_in[2];
    const float* W2a = (const float*)d_in[3];  const float* b2a = (const float*)d_in[4];
    const float* W1c = (const float*)d_in[5];  const float* b1c = (const float*)d_in[6];
    const float* W2c = (const float*)d_in[7];  const float* b2c = (const float*)d_in[8];
    const float* W1g = (const float*)d_in[9];  const float* b1g = (const float*)d_in[10];
    const float* W2g = (const float*)d_in[11]; const float* b2g = (const float*)d_in[12];
    const int* s1   = (const int*)d_in[13];
    const int* s2   = (const int*)d_in[14];
    const int* len1 = (const int*)d_in[15];
    const int* len2 = (const int*)d_in[16];

    char* ws = (char*)d_ws;
    size_t off = 0;
    auto alloc = [&](size_t bytes) { size_t o = off; off += (bytes + 255) & ~(size_t)255; return o; };

    bf16*  e1b    = (bf16*)(ws + alloc((size_t)ML * EMBP * 2));
    bf16*  e2b    = (bf16*)(ws + alloc((size_t)ML * EMBP * 2));
    bf16*  tbuf   = (bf16*)(ws + alloc((size_t)ML * HIDP * 2));   // MLP hidden / cmp hidden
    bf16*  h1     = (bf16*)(ws + alloc((size_t)ML * HIDP * 2));   // also cmp layer2 out
    bf16*  h2     = (bf16*)(ws + alloc((size_t)ML * HIDP * 2));
    float* scores = (float*)(ws + alloc((size_t)BBATCH * 65536 * 4)); // also fp32 cmp accum
    float* rm     = (float*)(ws + alloc((size_t)ML * 4));
    float* sb     = (float*)(ws + alloc((size_t)ML * 4));
    float* sa     = (float*)(ws + alloc((size_t)ML * 4));
    bf16*  eaT    = (bf16*)(ws + alloc((size_t)BBATCH * 65536 * 2));
    bf16*  ebeta  = (bf16*)(ws + alloc((size_t)BBATCH * 65536 * 2));
    bf16*  betas  = (bf16*)(ws + alloc((size_t)ML * EMBP * 2));
    bf16*  alphas = (bf16*)(ws + alloc((size_t)ML * EMBP * 2));
    bf16*  v1v2   = (bf16*)(ws + alloc((size_t)BBATCH * AGGKP * 2));
    bf16*  aggH   = (bf16*)(ws + alloc((size_t)BBATCH * HIDP * 2));
    bf16*  W1a_t  = (bf16*)(ws + alloc((size_t)256 * EMBP * 2));
    bf16*  W2a_t  = (bf16*)(ws + alloc((size_t)256 * HIDP * 2));
    bf16*  WcA_t  = (bf16*)(ws + alloc((size_t)256 * EMBP * 2));
    bf16*  WcB_t  = (bf16*)(ws + alloc((size_t)256 * EMBP * 2));
    bf16*  W2c_t  = (bf16*)(ws + alloc((size_t)256 * HIDP * 2));
    bf16*  W1g_t  = (bf16*)(ws + alloc((size_t)256 * AGGKP * 2));
    float* tmpF   = scores;      // reuse after softmax consumed
    bf16*  hC     = tbuf;        // reuse
    bf16*  oC     = h1;          // reuse

    auto cwgrid = [](int Kpad, int Npad) { return dim3((Kpad * Npad + 255) / 256); };

    // 1) weights -> transposed zero-padded bf16
    conv_weight<<<cwgrid(EMBP, 256), 256, 0, stream>>>(W1a, 0,   EMB, HID, W1a_t, EMBP, 256);
    conv_weight<<<cwgrid(HIDP, 256), 256, 0, stream>>>(W2a, 0,   HID, HID, W2a_t, HIDP, 256);
    conv_weight<<<cwgrid(EMBP, 256), 256, 0, stream>>>(W1c, 0,   EMB, HID, WcA_t, EMBP, 256);
    conv_weight<<<cwgrid(EMBP, 256), 256, 0, stream>>>(W1c, EMB, EMB, HID, WcB_t, EMBP, 256);
    conv_weight<<<cwgrid(HIDP, 256), 256, 0, stream>>>(W2c, 0,   HID, HID, W2c_t, HIDP, 256);
    conv_weight<<<cwgrid(AGGKP,256), 256, 0, stream>>>(W1g, 0,  AGGK, HID, W1g_t, AGGKP, 256);

    // 2) embedding gather
    gather_embed<<<dim3((unsigned)ML), 256, 0, stream>>>(emb, s1, e1b);
    gather_embed<<<dim3((unsigned)ML), 256, 0, stream>>>(emb, s2, e2b);

    // 3) attend MLP for both sequences: h = relu(relu(e@W1a+b1a)@W2a+b2a)
    gemm_wmma<false,true,true,false,true><<<dim3(1024,4,1),256,0,stream>>>(
        e1b, 0, W1a_t, 0, tbuf, 0, nullptr, b1a, HID, EMBP, EMBP, EMBP, HIDP);
    gemm_wmma<false,true,true,false,true><<<dim3(1024,4,1),256,0,stream>>>(
        tbuf, 0, W2a_t, 0, h1, 0, nullptr, b2a, HID, HIDP, HIDP, HIDP, HIDP);
    gemm_wmma<false,true,true,false,true><<<dim3(1024,4,1),256,0,stream>>>(
        e2b, 0, W1a_t, 0, tbuf, 0, nullptr, b1a, HID, EMBP, EMBP, EMBP, HIDP);
    gemm_wmma<false,true,true,false,true><<<dim3(1024,4,1),256,0,stream>>>(
        tbuf, 0, W2a_t, 0, h2, 0, nullptr, b2a, HID, HIDP, HIDP, HIDP, HIDP);

    // 4) scores: e[b] = h1[b] @ h2[b]^T  (BT path: "Wt" == h2, already N x K)
    gemm_wmma<true,false,false,false,true><<<dim3(2,4,BBATCH),256,0,stream>>>(
        h1, (long long)LSEQ*HIDP, h2, (long long)LSEQ*HIDP,
        scores, (long long)LSEQ*LSEQ, nullptr, nullptr, 0, HIDP, HIDP, HIDP, LSEQ);

    // 5) softmax statistics + normalized (transposed) attention weights
    row_max_sum<<<dim3((unsigned)ML),256,0,stream>>>(scores, len2, rm, sb);
    col_sum    <<<dim3((unsigned)ML),256,0,stream>>>(scores, rm, len1, sa);
    norm_write <<<dim3((unsigned)(BBATCH*65536/256)),256,0,stream>>>(
        scores, rm, sa, sb, len1, len2, ebeta, eaT);

    // 6) attended vectors: betas = e_beta @ e2 ; alphas = e_alpha^T @ e1   (B in K x N form)
    gemm_wmma<false,false,false,false,false><<<dim3(2,5,BBATCH),256,0,stream>>>(
        ebeta, (long long)65536, e2b, (long long)LSEQ*EMBP,
        betas, (long long)LSEQ*EMBP, nullptr, nullptr, 0, LSEQ, LSEQ, EMBP, EMBP);
    gemm_wmma<false,false,false,false,false><<<dim3(2,5,BBATCH),256,0,stream>>>(
        eaT, (long long)65536, e1b, (long long)LSEQ*EMBP,
        alphas, (long long)LSEQ*EMBP, nullptr, nullptr, 0, LSEQ, LSEQ, EMBP, EMBP);

    // 7) compare MLP, K=600 split as two K=320 passes (fp32 accumulate through tmpF)
    //    sequence 1: concat(e1, betas)
    gemm_wmma<true,false,false,false,true><<<dim3(1024,4,1),256,0,stream>>>(
        e1b, 0, WcA_t, 0, tmpF, 0, nullptr, nullptr, 0, EMBP, EMBP, EMBP, HIDP);
    gemm_wmma<false,true,true,true,true><<<dim3(1024,4,1),256,0,stream>>>(
        betas, 0, WcB_t, 0, hC, 0, tmpF, b1c, HID, EMBP, EMBP, EMBP, HIDP);
    gemm_wmma<false,true,true,false,true><<<dim3(1024,4,1),256,0,stream>>>(
        hC, 0, W2c_t, 0, oC, 0, nullptr, b2c, HID, HIDP, HIDP, HIDP, HIDP);
    masked_sum<<<dim3(BBATCH),256,0,stream>>>(oC, len1, v1v2, AGGKP, 0, HID, HIDP);

    //    sequence 2: concat(e2, alphas)  (pad cols of oC are zero -> Ncopy=216 also zeroes v1v2 tail)
    gemm_wmma<true,false,false,false,true><<<dim3(1024,4,1),256,0,stream>>>(
        e2b, 0, WcA_t, 0, tmpF, 0, nullptr, nullptr, 0, EMBP, EMBP, EMBP, HIDP);
    gemm_wmma<false,true,true,true,true><<<dim3(1024,4,1),256,0,stream>>>(
        alphas, 0, WcB_t, 0, hC, 0, tmpF, b1c, HID, EMBP, EMBP, EMBP, HIDP);
    gemm_wmma<false,true,true,false,true><<<dim3(1024,4,1),256,0,stream>>>(
        hC, 0, W2c_t, 0, oC, 0, nullptr, b2c, HID, HIDP, HIDP, HIDP, HIDP);
    masked_sum<<<dim3(BBATCH),256,0,stream>>>(oC, len2, v1v2, AGGKP, HID, 216, HIDP);

    // 8) aggregate: relu([v1|v2] @ W1g + b1g) then @ W2g + b2g
    gemm_wmma<false,true,true,false,true><<<dim3(4,4,1),256,0,stream>>>(
        v1v2, 0, W1g_t, 0, aggH, 0, nullptr, b1g, HID, AGGKP, AGGKP, AGGKP, HIDP);
    final_out<<<dim3(4),256,0,stream>>>(aggH, W2g, b2g, (float*)d_out);

    (void)in_sizes; (void)n_in; (void)out_size; (void)ws_size;
}
